// DeepSeekV3MoE_11269994184873
// MI455X (gfx1250) — matrix operand order repaired and checked by reading, not compiled, not run on hardware
//
#include <hip/hip_runtime.h>

// ---------------- problem constants ----------------
#define Hdim  1024
#define Idim  512
#define Edim  8
#define Tdim  2048       // B*S tokens
#define SIdim 1024       // shared expert intermediate
#define BKs   64         // K-slab per LDS stage
#define LDT   72         // padded LDS row stride (bf16): 144B, 16B-aligned, conflict-avoiding

// Async global->LDS DMA (CDNA5), guarded so compile never regresses.
#if defined(__has_builtin)
#if __has_builtin(__builtin_amdgcn_global_load_async_to_lds_b128) && \
    __has_builtin(__builtin_amdgcn_s_wait_asynccnt)
#define HAVE_ASYNC_LDS 1
#endif
#endif
#ifndef HAVE_ASYNC_LDS
#define HAVE_ASYNC_LDS 0
#endif

typedef __attribute__((ext_vector_type(16))) __bf16 v16bf;
typedef __attribute__((ext_vector_type(8)))  float  v8f;
typedef __attribute__((ext_vector_type(4)))  int    v4i;

#define WMMA_BF16(a, b, c) \
    __builtin_amdgcn_wmma_f32_16x16x32_bf16(false, (a), false, (b), (short)0, (c), false, false)

// ---------------- helpers ----------------
static __device__ __forceinline__ unsigned short f32_to_bf16(float f) {
    unsigned int u = __float_as_uint(f);
    u += 0x7FFFu + ((u >> 16) & 1u);   // round-to-nearest-even
    return (unsigned short)(u >> 16);
}

static __device__ __forceinline__ float silu_f(float x) {
    return x / (1.0f + expf(-x));
}

// 16B global->LDS copy: async DMA when available, ld/st fallback otherwise.
static __device__ __forceinline__ void cp16(const unsigned short* g, unsigned short* l) {
#if HAVE_ASYNC_LDS
    __builtin_amdgcn_global_load_async_to_lds_b128(
        (__attribute__((address_space(1))) v4i*)(const v4i*)g,
        (__attribute__((address_space(3))) v4i*)(v4i*)l, 0, 0);
#else
    *(uint4*)l = *(const uint4*)g;
#endif
}
static __device__ __forceinline__ void async_wait_all() {
#if HAVE_ASYNC_LDS
    __builtin_amdgcn_s_wait_asynccnt(0);
#endif
}

// A fragment: 16x32 bf16, wave32 layout (lanes 0-15: M rows, K-half 0; lanes 16-31: K-half 1)
static __device__ __forceinline__ v16bf lds_a_frag(const unsigned short* __restrict__ At,
                                                   int m_off, int kbase, int lane) {
    const int m  = m_off + (lane & 15);
    const int hi = (lane >> 4) & 1;
    union { v16bf v; unsigned int u[8]; } f;
    const unsigned short* row = At + m * LDT + kbase;
#pragma unroll
    for (int v = 0; v < 4; ++v)
        f.u[v] = *(const unsigned int*)(row + 16 * hi + 2 * v);
#pragma unroll
    for (int v = 0; v < 4; ++v)
        f.u[4 + v] = *(const unsigned int*)(row + 16 * hi + 8 + 2 * v);
    return f.v;
}

// B fragment: 32x16 bf16 stored as [N][K] rows; lanes 0-15: K=0..15, lanes 16-31: K=16..31
static __device__ __forceinline__ v16bf lds_b_frag(const unsigned short* __restrict__ Bt,
                                                   int n_off, int kbase, int lane) {
    const int n  = n_off + (lane & 15);
    const int hi = (lane >> 4) & 1;
    union { v16bf v; unsigned int u[8]; } f;
    const unsigned short* row = Bt + n * LDT + kbase;
#pragma unroll
    for (int v = 0; v < 8; ++v)
        f.u[v] = *(const unsigned int*)(row + 16 * hi + 2 * v);
    return f.v;
}

// ---------------- small kernels ----------------
__global__ void moe_zero_counters(int* __restrict__ cnt, float* __restrict__ sum_w) {
    int i = threadIdx.x;
    if (i < Edim) { cnt[i] = 0; sum_w[i] = 0.0f; }
}

__global__ void moe_cvt_bf16(const float* __restrict__ in, unsigned short* __restrict__ out, int n4) {
    int i = blockIdx.x * 256 + threadIdx.x;
    if (i >= n4) return;
    float4 f = ((const float4*)in)[i];
    uint2 o;
    o.x = (unsigned int)f32_to_bf16(f.x) | ((unsigned int)f32_to_bf16(f.y) << 16);
    o.y = (unsigned int)f32_to_bf16(f.z) | ((unsigned int)f32_to_bf16(f.w) << 16);
    ((uint2*)out)[i] = o;
}

// One wave32 per token: logits over 8 experts, sigmoid, top-2 (lowest-index tie-break),
// append (token -> expert) assignments; accumulate per-expert weight sums & counts.
__global__ __launch_bounds__(256) void moe_router(
    const float* __restrict__ x, const float* __restrict__ gate_w,
    const float* __restrict__ lb_bias,
    int* __restrict__ cnt, float* __restrict__ sum_w, int* __restrict__ list) {
    const int lane = threadIdx.x & 31;
    const int wave = threadIdx.x >> 5;
    const int t = blockIdx.x * 8 + wave;
    if (t >= Tdim) return;
    float acc[Edim];
#pragma unroll
    for (int e = 0; e < Edim; ++e) acc[e] = 0.0f;
    const float* xr = x + (size_t)t * Hdim;
    for (int k = lane; k < Hdim; k += 32) {
        float xv = xr[k];
#pragma unroll
        for (int e = 0; e < Edim; ++e) acc[e] += xv * gate_w[e * Hdim + k];
    }
#pragma unroll
    for (int e = 0; e < Edim; ++e) {
#pragma unroll
        for (int off = 16; off > 0; off >>= 1) acc[e] += __shfl_xor(acc[e], off, 32);
    }
    if (lane == 0) {
        float s[Edim];
#pragma unroll
        for (int e = 0; e < Edim; ++e)
            s[e] = 1.0f / (1.0f + expf(-(acc[e] + lb_bias[e])));
        int i0 = 0;
#pragma unroll
        for (int e = 1; e < Edim; ++e) if (s[e] > s[i0]) i0 = e;
        int i1 = (i0 == 0) ? 1 : 0;
#pragma unroll
        for (int e = 0; e < Edim; ++e) if (e != i0 && s[e] > s[i1]) i1 = e;
        float denom = s[i0] + s[i1] + 1e-8f;
        float w0 = s[i0] / denom, w1 = s[i1] / denom;
        int p0 = atomicAdd(&cnt[i0], 1);
        list[i0 * Tdim + p0] = t;
        __hip_atomic_fetch_add(&sum_w[i0], w0, __ATOMIC_RELAXED, __HIP_MEMORY_SCOPE_AGENT);
        int p1 = atomicAdd(&cnt[i1], 1);
        list[i1 * Tdim + p1] = t;
        __hip_atomic_fetch_add(&sum_w[i1], w1, __ATOMIC_RELAXED, __HIP_MEMORY_SCOPE_AGENT);
    }
}

__global__ void moe_wmean(const int* __restrict__ cnt, const float* __restrict__ sum_w,
                          float* __restrict__ wmean) {
    int e = threadIdx.x;
    if (e < Edim) {
        int c = cnt[e];
        wmean[e] = sum_w[e] / (float)(c > 0 ? c : 1);
    }
}

// ---------------- fused gate+up GEMM ----------------
// Block tile 128(M) x 64(N) x 64(K-slab), 8 waves as 4x2, per-wave 32x32.
// ROUTED: A rows gathered through per-expert token list (clamped, never predicated).
// HSTR: row stride of the hidden output (Idim routed / SIdim shared).
template <bool ROUTED, int HSTR>
__global__ __launch_bounds__(256) void moe_gateup(
    const unsigned short* __restrict__ xbf,
    const unsigned short* __restrict__ gw, const unsigned short* __restrict__ uw,
    unsigned short* __restrict__ hidden,
    const int* __restrict__ list, const int* __restrict__ cnt) {
    __shared__ unsigned short At[2][128 * LDT];
    __shared__ unsigned short Bg[2][64 * LDT];
    __shared__ unsigned short Bu[2][64 * LDT];

    const int e = ROUTED ? blockIdx.z : 0;
    int ne = Tdim;
    if constexpr (ROUTED) {
        ne = cnt[e];
        if ((int)blockIdx.x * 128 >= ne) return;
    }
    const int m0 = blockIdx.x * 128, n0 = blockIdx.y * 64;
    const int tid  = threadIdx.x;
    const int lane = tid & 31, wave = tid >> 5;
    const int m_w = (wave >> 1) * 32, n_w = (wave & 1) * 32;

    // loader assignment: A row per thread-pair (two 32-elem halves), B row per thread-quad
    const int arow = tid >> 1, ahalf = (tid & 1) * 32;
    int atok = m0 + arow;
    if constexpr (ROUTED) atok = list[e * Tdim + (m0 + arow < ne ? m0 + arow : ne - 1)];
    const unsigned short* ag = xbf + (size_t)atok * Hdim + ahalf;
    const int brow = tid >> 2, bq = (tid & 3) * 16;
    const size_t wofs = (ROUTED ? (size_t)e * Idim * Hdim : 0) + (size_t)(n0 + brow) * Hdim + bq;
    const unsigned short* gg = gw + wofs;
    const unsigned short* ug = uw + wofs;
    unsigned short* al[2] = { &At[0][arow * LDT + ahalf], &At[1][arow * LDT + ahalf] };
    unsigned short* gl[2] = { &Bg[0][brow * LDT + bq],    &Bg[1][brow * LDT + bq] };
    unsigned short* ul[2] = { &Bu[0][brow * LDT + bq],    &Bu[1][brow * LDT + bq] };

    auto issue = [&](int kk, int buf) {
        const unsigned short* a = ag + kk;
        cp16(a,      al[buf]);      cp16(a + 8,  al[buf] + 8);
        cp16(a + 16, al[buf] + 16); cp16(a + 24, al[buf] + 24);
        const unsigned short* g = gg + kk;
        cp16(g, gl[buf]);           cp16(g + 8, gl[buf] + 8);
        const unsigned short* u = ug + kk;
        cp16(u, ul[buf]);           cp16(u + 8, ul[buf] + 8);
    };

    v8f cg[2][2] = {}, cu[2][2] = {};
    constexpr int NSLAB = Hdim / BKs;
    issue(0, 0);
    for (int s = 0; s < NSLAB; ++s) {
        const int buf = s & 1;
        async_wait_all();
        __syncthreads();                       // slab s resident; all waves past slab s-1
        if (s + 1 < NSLAB) issue((s + 1) * BKs, buf ^ 1);  // DMA next slab during compute
#pragma unroll
        for (int ks = 0; ks < 2; ++ks) {
            const int kb = ks * 32;
            v16bf a0 = lds_a_frag(At[buf], m_w,      kb, lane);
            v16bf a1 = lds_a_frag(At[buf], m_w + 16, kb, lane);
            v16bf b0 = lds_b_frag(Bg[buf], n_w,      kb, lane);
            v16bf b1 = lds_b_frag(Bg[buf], n_w + 16, kb, lane);
            cg[0][0] = WMMA_BF16(a0, b0, cg[0][0]);
            cg[0][1] = WMMA_BF16(a0, b1, cg[0][1]);
            cg[1][0] = WMMA_BF16(a1, b0, cg[1][0]);
            cg[1][1] = WMMA_BF16(a1, b1, cg[1][1]);
            b0 = lds_b_frag(Bu[buf], n_w,      kb, lane);
            b1 = lds_b_frag(Bu[buf], n_w + 16, kb, lane);
            cu[0][0] = WMMA_BF16(a0, b0, cu[0][0]);
            cu[0][1] = WMMA_BF16(a0, b1, cu[0][1]);
            cu[1][0] = WMMA_BF16(a1, b0, cu[1][0]);
            cu[1][1] = WMMA_BF16(a1, b1, cu[1][1]);
        }
    }
    const int nl = lane & 15, hi = lane >> 4;
#pragma unroll
    for (int mi = 0; mi < 2; ++mi)
#pragma unroll
        for (int ni = 0; ni < 2; ++ni)
#pragma unroll
            for (int r = 0; r < 8; ++r) {
                int gm = m0 + m_w + mi * 16 + 8 * hi + r;
                if (!ROUTED || gm < ne) {
                    size_t rowp = ((size_t)(ROUTED ? e * Tdim : 0) + gm) * HSTR
                                + n0 + n_w + ni * 16 + nl;
                    hidden[rowp] = f32_to_bf16(silu_f(cg[mi][ni][r]) * cu[mi][ni][r]);
                }
            }
}

// ---------------- down projection GEMM ----------------
// Block tile 128x64xK-slab 64. ROUTED: scaled atomic scatter; else plain store (inits d_out).
template <bool ROUTED, int KD>
__global__ __launch_bounds__(256) void moe_down(
    const unsigned short* __restrict__ hid,
    const unsigned short* __restrict__ dw,
    float* __restrict__ out,
    const int* __restrict__ list, const int* __restrict__ cnt,
    const float* __restrict__ wmean) {
    __shared__ unsigned short At[2][128 * LDT];
    __shared__ unsigned short Bt[2][64 * LDT];

    const int e = ROUTED ? blockIdx.z : 0;
    int ne = Tdim;
    if constexpr (ROUTED) {
        ne = cnt[e];
        if ((int)blockIdx.x * 128 >= ne) return;
    }
    const int m0 = blockIdx.x * 128, n0 = blockIdx.y * 64;
    const int tid  = threadIdx.x;
    const int lane = tid & 31, wave = tid >> 5;
    const int m_w = (wave >> 1) * 32, n_w = (wave & 1) * 32;

    const int arow = tid >> 1, ahalf = (tid & 1) * 32;
    int ar = m0 + arow;
    if constexpr (ROUTED) { if (ar >= ne) ar = ne - 1; }
    const unsigned short* ag = hid + ((size_t)(ROUTED ? e * Tdim : 0) + ar) * KD + ahalf;
    const int brow = tid >> 2, bq = (tid & 3) * 16;
    const unsigned short* bg = dw + (ROUTED ? (size_t)e * Hdim * KD : 0)
                                  + (size_t)(n0 + brow) * KD + bq;
    unsigned short* al[2] = { &At[0][arow * LDT + ahalf], &At[1][arow * LDT + ahalf] };
    unsigned short* bl[2] = { &Bt[0][brow * LDT + bq],    &Bt[1][brow * LDT + bq] };

    auto issue = [&](int kk, int buf) {
        const unsigned short* a = ag + kk;
        cp16(a,      al[buf]);      cp16(a + 8,  al[buf] + 8);
        cp16(a + 16, al[buf] + 16); cp16(a + 24, al[buf] + 24);
        const unsigned short* b = bg + kk;
        cp16(b, bl[buf]);           cp16(b + 8, bl[buf] + 8);
    };

    v8f c[2][2] = {};
    constexpr int NSLAB = KD / BKs;
    issue(0, 0);
    for (int s = 0; s < NSLAB; ++s) {
        const int buf = s & 1;
        async_wait_all();
        __syncthreads();
        if (s + 1 < NSLAB) issue((s + 1) * BKs, buf ^ 1);
#pragma unroll
        for (int ks = 0; ks < 2; ++ks) {
            const int kb = ks * 32;
            v16bf a0 = lds_a_frag(At[buf], m_w,      kb, lane);
            v16bf a1 = lds_a_frag(At[buf], m_w + 16, kb, lane);
            v16bf b0 = lds_b_frag(Bt[buf], n_w,      kb, lane);
            v16bf b1 = lds_b_frag(Bt[buf], n_w + 16, kb, lane);
            c[0][0] = WMMA_BF16(a0, b0, c[0][0]);
            c[0][1] = WMMA_BF16(a0, b1, c[0][1]);
            c[1][0] = WMMA_BF16(a1, b0, c[1][0]);
            c[1][1] = WMMA_BF16(a1, b1, c[1][1]);
        }
    }
    const int nl = lane & 15, hi = lane >> 4;
    const float sc = ROUTED ? wmean[e] : 1.0f;
#pragma unroll
    for (int mi = 0; mi < 2; ++mi)
#pragma unroll
        for (int ni = 0; ni < 2; ++ni)
#pragma unroll
            for (int r = 0; r < 8; ++r) {
                int gm = m0 + m_w + mi * 16 + 8 * hi + r;
                int col = n0 + n_w + ni * 16 + nl;
                if constexpr (ROUTED) {
                    if (gm < ne) {
                        int tok = list[e * Tdim + gm];
                        __hip_atomic_fetch_add(out + (size_t)tok * Hdim + col,
                                               c[mi][ni][r] * sc,
                                               __ATOMIC_RELAXED, __HIP_MEMORY_SCOPE_AGENT);
                    }
                } else {
                    out[(size_t)gm * Hdim + col] = c[mi][ni][r];
                }
            }
}

// ---------------- launcher ----------------
extern "C" void kernel_launch(void* const* d_in, const int* in_sizes, int n_in,
                              void* d_out, int out_size, void* d_ws, size_t ws_size,
                              hipStream_t stream) {
    const float* x      = (const float*)d_in[0];
    const float* gate_w = (const float*)d_in[1];
    const float* lb     = (const float*)d_in[2];
    const float* egw    = (const float*)d_in[3];
    const float* euw    = (const float*)d_in[4];
    const float* edw    = (const float*)d_in[5];
    const float* sgw    = (const float*)d_in[6];
    const float* suw    = (const float*)d_in[7];
    const float* sdw    = (const float*)d_in[8];
    float* out = (float*)d_out;

    char* ws = (char*)d_ws;
    size_t off = 0;
    auto alloc = [&](size_t bytes) -> void* {
        void* p = ws + off;
        off += (bytes + 255) & ~(size_t)255;
        return p;
    };
    unsigned short* xbf   = (unsigned short*)alloc((size_t)Tdim * Hdim * 2);
    unsigned short* egwb  = (unsigned short*)alloc((size_t)Edim * Idim * Hdim * 2);
    unsigned short* euwb  = (unsigned short*)alloc((size_t)Edim * Idim * Hdim * 2);
    unsigned short* edwb  = (unsigned short*)alloc((size_t)Edim * Hdim * Idim * 2);
    unsigned short* sgwb  = (unsigned short*)alloc((size_t)SIdim * Hdim * 2);
    unsigned short* suwb  = (unsigned short*)alloc((size_t)SIdim * Hdim * 2);
    unsigned short* sdwb  = (unsigned short*)alloc((size_t)Hdim * SIdim * 2);
    unsigned short* hid_r = (unsigned short*)alloc((size_t)Edim * Tdim * Idim * 2);
    unsigned short* hid_s = (unsigned short*)alloc((size_t)Tdim * SIdim * 2);
    int*   list  = (int*)alloc((size_t)Edim * Tdim * 4);
    int*   cnt   = (int*)alloc(Edim * 4);
    float* sum_w = (float*)alloc(Edim * 4);
    float* wmean = (float*)alloc(Edim * 4);

    moe_zero_counters<<<1, 32, 0, stream>>>(cnt, sum_w);

    auto cvt = [&](const float* src, unsigned short* dst, size_t n) {
        int n4 = (int)(n / 4);
        moe_cvt_bf16<<<(n4 + 255) / 256, 256, 0, stream>>>(src, dst, n4);
    };
    cvt(x,   xbf,  (size_t)Tdim * Hdim);
    cvt(egw, egwb, (size_t)Edim * Idim * Hdim);
    cvt(euw, euwb, (size_t)Edim * Idim * Hdim);
    cvt(edw, edwb, (size_t)Edim * Hdim * Idim);
    cvt(sgw, sgwb, (size_t)SIdim * Hdim);
    cvt(suw, suwb, (size_t)SIdim * Hdim);
    cvt(sdw, sdwb, (size_t)Hdim * SIdim);

    moe_router<<<Tdim / 8, 256, 0, stream>>>(x, gate_w, lb, cnt, sum_w, list);
    moe_wmean<<<1, 32, 0, stream>>>(cnt, sum_w, wmean);

    // routed experts: gathered gate+up -> hidden
    moe_gateup<true, Idim><<<dim3(Tdim / 128, Idim / 64, Edim), 256, 0, stream>>>(
        xbf, egwb, euwb, hid_r, list, cnt);

    // shared expert: dense gate+up, then down (initializes d_out with plain stores)
    moe_gateup<false, SIdim><<<dim3(Tdim / 128, SIdim / 64), 256, 0, stream>>>(
        xbf, sgwb, suwb, hid_s, nullptr, nullptr);
    moe_down<false, SIdim><<<dim3(Tdim / 128, Hdim / 64), 256, 0, stream>>>(
        hid_s, sdwb, out, nullptr, nullptr, nullptr);

    // routed experts: down, scaled by per-expert mean weight, atomic scatter into d_out
    moe_down<true, Idim><<<dim3(Tdim / 128, Hdim / 64, Edim), 256, 0, stream>>>(
        hid_r, edwb, out, list, cnt, wmean);

    (void)in_sizes; (void)n_in; (void)out_size; (void)ws_size;
}
